// TopKSAE_70085276336587
// MI455X (gfx1250) — compile-verified, matrix operand order checked by reading, not run on hardware
//
#include <hip/hip_runtime.h>
#include <hip/hip_bf16.h>

// ---------------------------------------------------------------------------
// Problem constants (fixed by the reference)
// ---------------------------------------------------------------------------
#define BATCH   16384
#define DIM     1536      // INPUT_DIM
#define HID     12288     // HIDDEN_DIM
#define KEEP    32        // top-k

typedef __bf16         bf16_t;
typedef __bf16         v16bf  __attribute__((ext_vector_type(16)));
typedef float          v8f    __attribute__((ext_vector_type(8)));
typedef unsigned int   u32x4  __attribute__((ext_vector_type(4)));
typedef int            i32x4  __attribute__((ext_vector_type(4)));
typedef int            i32x8  __attribute__((ext_vector_type(8)));
typedef unsigned short u16;

// f32 -> bf16 round-to-nearest-even (bit pattern)
__device__ __forceinline__ u16 f32_to_bf16_rne(float f) {
    unsigned int u = __float_as_uint(f);
    unsigned int r = u + 0x7FFFu + ((u >> 16) & 1u);
    return (u16)(r >> 16);
}

// monotone key: larger float -> larger unsigned key
__device__ __forceinline__ unsigned int f32_key(float f) {
    unsigned int u = __float_as_uint(f);
    return (u & 0x80000000u) ? ~u : (u | 0x80000000u);
}

// Raw LDS byte address of a __shared__ object: per the CDNA5 aperture rules
// (ISA 10.2) a generic LDS address is {SHARED_BASE[63:32], lds_addr[31:0]},
// so the low 32 bits of the flat pointer are the LDS byte address.
__device__ __forceinline__ unsigned lds_addr_of(const void* p) {
    return (unsigned)(unsigned long long)p;
}

// ---------------------------------------------------------------------------
// TDM: post a 2D-tile DMA (Global -> LDS) via the Tensor Data Mover.
//   tile = tile_dim1 rows x 32 contiguous bf16 (64B) per row,
//   row stride in memory = DIM elements, packed row-major into LDS.
// Descriptor layout per CDNA5 ISA §8.3 (group 0) / §8.4 (group 1).
// Tracked with TENSORcnt; issue from one wave, EXEC is ignored.
// ---------------------------------------------------------------------------
__device__ __forceinline__ void tdm_load_tile(unsigned lds_byte,
                                              const void* gtile,
                                              unsigned tensor_rows) {
    unsigned long long ga = (unsigned long long)gtile;
    u32x4 g0;
    g0.x = 1u;                                              // count=1, user D#
    g0.y = lds_byte;                                        // lds_addr [63:32]
    g0.z = (unsigned)ga;                                    // global_addr lo
    g0.w = ((unsigned)(ga >> 32) & 0x01FFFFFFu)             // global_addr hi
         | 0x80000000u;                                     // type=2 ("image")
    i32x8 g1;
    g1[0] = 0x00010000;                                     // data_size=1 -> 2B; wg_mask=0
    g1[1] = (int)(((unsigned)DIM & 0xFFFFu) << 16);         // tensor_dim0[15:0] @ [63:48]
    g1[2] = (int)((tensor_rows & 0xFFFFu) << 16);           // tensor_dim1[15:0] @ [95:80]
    g1[3] = (int)(32u << 16);                               // tile_dim0=32 @ [127:112]
    g1[4] = 128;                                            // tile_dim1=128 @ [143:128]
    g1[5] = DIM;                                            // tensor_dim0_stride[31:0]
    g1[6] = 0;                                              // stride hi / dim1_stride
    g1[7] = 0;
    i32x4 z4 = {0, 0, 0, 0};                                // 2D: groups 2/3 unused
#if __clang_major__ >= 23
    i32x8 z8 = {0, 0, 0, 0, 0, 0, 0, 0};
    __builtin_amdgcn_tensor_load_to_lds(g0, g1, z4, z4, z8, 0);
#else
    __builtin_amdgcn_tensor_load_to_lds(g0, g1, z4, z4, 0);
#endif
}

// ---------------------------------------------------------------------------
// Kernel 1: xc_bf16 = bf16(x - b_dec)          (BATCH x DIM)
// ---------------------------------------------------------------------------
__global__ __launch_bounds__(256) void sae_cvt_x(const float* __restrict__ x,
                                                 const float* __restrict__ b_dec,
                                                 u16* __restrict__ out) {
    unsigned int i = blockIdx.x * 256u + threadIdx.x;   // < BATCH*DIM
    unsigned int col = i % DIM;
    out[i] = f32_to_bf16_rne(x[i] - b_dec[col]);
}

// ---------------------------------------------------------------------------
// Kernel 2: W_enc -> bf16                       (HID x DIM)
// ---------------------------------------------------------------------------
__global__ __launch_bounds__(256) void sae_cvt_w(const float* __restrict__ w,
                                                 u16* __restrict__ out) {
    unsigned int i = blockIdx.x * 256u + threadIdx.x;
    out[i] = f32_to_bf16_rne(w[i]);
}

// ---------------------------------------------------------------------------
// Kernel 3: encoder GEMM  C[B,H] = Xc(bf16) @ Wenc(bf16)^T + b_enc
// C[m,n] = sum_k A[m,k]*B[n,k]  -- both operands contiguous along k.
// Block tile 128x128, BK=32 (one WMMA K-slab), 256 threads = 8 waves.
// Wave grid 2x4: each wave computes a 64x32 tile = 4x2 WMMA 16x16 tiles.
// Tiles are staged by the Tensor Data Mover, double-buffered in LDS so the
// DMA of slab k+1 overlaps the WMMAs on slab k; one barrier per K-step.
// ---------------------------------------------------------------------------
#define BM 128
#define BN 128
#define BK 32

__global__ __launch_bounds__(256) void sae_enc_gemm(const u16* __restrict__ Abf,
                                                    const u16* __restrict__ Bbf,
                                                    const float* __restrict__ b_enc,
                                                    float* __restrict__ C) {
    __shared__ u16 As[2][BM * BK];   // 2 x 8 KB
    __shared__ u16 Bs[2][BN * BK];   // 2 x 8 KB

    const int tid  = threadIdx.x;
    const int lane = tid & 31;
    const int wave = tid >> 5;        // 0..7
    const int wm   = wave >> 2;       // 0..1  -> M offset wm*64
    const int wn   = wave & 3;        // 0..3  -> N offset wn*32
    const int bm   = blockIdx.y * BM; // batch rows
    const int bn   = blockIdx.x * BN; // hidden cols

    const int lr   = lane & 15;           // row/col within 16x16 frag
    const int half = (lane >> 4) << 4;    // K-half select (0 or 16)

    const u16* Arows = Abf + (size_t)bm * DIM;
    const u16* Brows = Bbf + (size_t)bn * DIM;
    const unsigned asb[2] = { lds_addr_of(&As[0][0]), lds_addr_of(&As[1][0]) };
    const unsigned bsb[2] = { lds_addr_of(&Bs[0][0]), lds_addr_of(&Bs[1][0]) };

    v8f acc[4][2];
#pragma unroll
    for (int mi = 0; mi < 4; ++mi)
#pragma unroll
        for (int ni = 0; ni < 2; ++ni)
            acc[mi][ni] = (v8f){0.f, 0.f, 0.f, 0.f, 0.f, 0.f, 0.f, 0.f};

    // prologue: post DMA for K-slab 0 into buffer 0
    if (wave == 0) {
        tdm_load_tile(asb[0], Arows, BATCH);
        tdm_load_tile(bsb[0], Brows, HID);
    }

    int cur = 0;
    for (int k0 = 0; k0 < DIM; k0 += BK) {
        if (wave == 0)
            __builtin_amdgcn_s_wait_tensorcnt(0);   // slab `cur` landed in LDS
        __syncthreads();                            // publish slab; all waves past
                                                    // their reads of slab cur^1
        if (wave == 0 && (k0 + BK) < DIM) {         // post DMA for next slab
            tdm_load_tile(asb[cur ^ 1], Arows + (k0 + BK), BATCH);
            tdm_load_tile(bsb[cur ^ 1], Brows + (k0 + BK), HID);
        }

        const u16* Ac = &As[cur][0];
        const u16* Bc = &Bs[cur][0];

        // fragments: lane holds 16 contiguous bf16 of its row (K-half selected)
        v16bf afrag[4], bfrag[2];
#pragma unroll
        for (int mi = 0; mi < 4; ++mi)
            afrag[mi] = *(const v16bf*)&Ac[(wm * 64 + mi * 16 + lr) * BK + half];
#pragma unroll
        for (int ni = 0; ni < 2; ++ni)
            bfrag[ni] = *(const v16bf*)&Bc[(wn * 32 + ni * 16 + lr) * BK + half];

#pragma unroll
        for (int mi = 0; mi < 4; ++mi)
#pragma unroll
            for (int ni = 0; ni < 2; ++ni)
                acc[mi][ni] = __builtin_amdgcn_wmma_f32_16x16x32_bf16(
                    /*neg_a=*/false, afrag[mi],
                    /*neg_b=*/false, bfrag[ni],
                    /*c_mod=*/(short)0, acc[mi][ni],
                    /*reuse_a=*/false, /*reuse_b=*/false);
        cur ^= 1;
    }

    // epilogue: C/D layout — lanes 0..15 hold N=lane, M=i; lanes 16..31 hold M=8+i
    const int mofs = (lane >> 4) * 8;
    const int ncol = lane & 15;
#pragma unroll
    for (int mi = 0; mi < 4; ++mi) {
#pragma unroll
        for (int ni = 0; ni < 2; ++ni) {
            const int col = bn + wn * 32 + ni * 16 + ncol;
            const float bias = b_enc[col];
#pragma unroll
            for (int i = 0; i < 8; ++i) {
                const int row = bm + wm * 64 + mi * 16 + mofs + i;
                C[(size_t)row * HID + col] = acc[mi][ni][i] + bias;
            }
        }
    }
}

// ---------------------------------------------------------------------------
// Kernel 4: per-row top-K threshold (radix binary search on monotone keys),
// in-place sparsify + ReLU of the pre-activation row (which lives in d_out),
// and record the (idx, val) survivors for the decoder.
// One workgroup (256 threads) per row; row keys staged in 48 KB LDS.
// ---------------------------------------------------------------------------
__global__ __launch_bounds__(256) void sae_topk(float* __restrict__ act,  // (B,H) in-place
                                                float* __restrict__ tv,   // (B,KEEP)
                                                int*   __restrict__ ti) { // (B,KEEP)
    __shared__ unsigned int skey[HID];    // 48 KB
    __shared__ int scnt;
    __shared__ int sclaim;

    const int row = blockIdx.x;
    float* p = act + (size_t)row * HID;
    const int tid = threadIdx.x;

    // stage monotone keys in LDS
    for (int j = tid; j < HID; j += 256)
        skey[j] = f32_key(p[j]);
    if (tid == 0) sclaim = 0;
    __syncthreads();

    // binary search: largest T with count(key >= T) >= KEEP
    unsigned int lo = 0u, hi = 0xFFFFFFFFu;
    for (int it = 0; it < 32; ++it) {
        if (lo >= hi) break;
        const unsigned int mid = lo + ((hi - lo) >> 1) + 1u;
        if (tid == 0) scnt = 0;
        __syncthreads();
        int c = 0;
        for (int j = tid; j < HID; j += 256)
            c += (skey[j] >= mid) ? 1 : 0;
        atomicAdd(&scnt, c);
        __syncthreads();
        const int total = scnt;
        __syncthreads();
        if (total >= KEEP) lo = mid; else hi = mid - 1u;
    }
    const unsigned int thresh = lo;

    // rewrite row: survivors -> relu(val), everything else -> 0; claim slots
    for (int j = tid; j < HID; j += 256) {
        const float v = p[j];
        const bool keep = (skey[j] >= thresh);
        const float out = keep ? fmaxf(v, 0.f) : 0.f;
        p[j] = out;
        if (keep) {
            const int slot = atomicAdd(&sclaim, 1);
            if (slot < KEEP) {
                ti[row * KEEP + slot] = j;
                tv[row * KEEP + slot] = out;
            }
        }
    }
}

// ---------------------------------------------------------------------------
// Kernel 5: sparse decode  x_hat[row,:] = sum_k tv[k] * W_dec[:, ti[k]] + b_dec
// One workgroup per row; W_dec (75 MB) is L2-resident so the strided column
// gathers are served from L2.
// ---------------------------------------------------------------------------
__global__ __launch_bounds__(256) void sae_dec(const float* __restrict__ tv,
                                               const int*   __restrict__ ti,
                                               const float* __restrict__ W_dec,
                                               const float* __restrict__ b_dec,
                                               float* __restrict__ xhat) {
    __shared__ float sv[KEEP];
    __shared__ int   si[KEEP];
    const int row = blockIdx.x;
    if (threadIdx.x < KEEP) {
        sv[threadIdx.x] = tv[row * KEEP + threadIdx.x];
        si[threadIdx.x] = ti[row * KEEP + threadIdx.x];
    }
    __syncthreads();
    for (int d = threadIdx.x; d < DIM; d += 256) {
        float accum = b_dec[d];
        const float* wrow = W_dec + (size_t)d * HID;
#pragma unroll
        for (int k = 0; k < KEEP; ++k)
            accum += sv[k] * wrow[si[k]];
        xhat[(size_t)row * DIM + d] = accum;
    }
}

// ---------------------------------------------------------------------------
// Host launcher
// ---------------------------------------------------------------------------
extern "C" void kernel_launch(void* const* d_in, const int* in_sizes, int n_in,
                              void* d_out, int out_size, void* d_ws, size_t ws_size,
                              hipStream_t stream) {
    const float* x     = (const float*)d_in[0];  // (B, D)
    const float* W_enc = (const float*)d_in[1];  // (H, D)
    const float* b_enc = (const float*)d_in[2];  // (H)
    const float* W_dec = (const float*)d_in[3];  // (D, H)
    const float* b_dec = (const float*)d_in[4];  // (D)

    float* out    = (float*)d_out;
    float* xhat   = out;                                  // (B, D)
    float* sparse = out + (size_t)BATCH * DIM;            // (B, H)

    char* ws = (char*)d_ws;
    u16*   Xbf = (u16*)ws;                                            // 50.3 MB
    u16*   Wbf = (u16*)(ws + (size_t)BATCH * DIM * sizeof(u16));      // 37.7 MB
    float* tv  = (float*)(ws + (size_t)BATCH * DIM * sizeof(u16)
                             + (size_t)HID * DIM * sizeof(u16));      // 2 MB
    int*   ti  = (int*)((char*)tv + (size_t)BATCH * KEEP * sizeof(float));

    // 1) xc = bf16(x - b_dec)
    sae_cvt_x<<<(BATCH * DIM) / 256, 256, 0, stream>>>(x, b_dec, Xbf);
    // 2) W_enc -> bf16
    sae_cvt_w<<<(HID * DIM) / 256, 256, 0, stream>>>(W_enc, Wbf);
    // 3) pre_act = Xc @ Wenc^T + b_enc  (written straight into sparse region)
    dim3 ggrid(HID / BN, BATCH / BM);   // (96, 128)
    sae_enc_gemm<<<ggrid, 256, 0, stream>>>(Xbf, Wbf, b_enc, sparse);
    // 4) per-row top-K sparsify + ReLU (in place) + survivor lists
    sae_topk<<<BATCH, 256, 0, stream>>>(sparse, tv, ti);
    // 5) sparse decode
    sae_dec<<<BATCH, 256, 0, stream>>>(tv, ti, W_dec, b_dec, xhat);
}
